// CrossAttention_14345190769330
// MI455X (gfx1250) — matrix operand order repaired and checked by reading, not compiled, not run on hardware
//
#include <hip/hip_runtime.h>
#include <hip/hip_bf16.h>

// ---------------------------------------------------------------------------
// CDNA5 (gfx1250) cross-attention, bf16 WMMA pipeline, f32 accumulate.
// ---------------------------------------------------------------------------

typedef __bf16 bf16_t;
typedef __attribute__((ext_vector_type(8)))  __bf16 v8bf;
typedef __attribute__((ext_vector_type(16))) __bf16 v16bf;
typedef __attribute__((ext_vector_type(8)))  float  v8f;

#define B_    4
#define N_    2048
#define M_    2048
#define QD_   1024
#define H_    8
#define D_    64
#define INNER_ 512
#define ROWS_ (B_ * N_)   // 8192

static __device__ __forceinline__ v16bf combine8(v8bf lo, v8bf hi) {
  v16bf r;
#pragma unroll
  for (int i = 0; i < 8; ++i) { r[i] = lo[i]; r[i + 8] = hi[i]; }
  return r;
}

// A-fragment (16x32 bf16), matrix row-major [row][ld], k contiguous.
// lane: row = base + (l&15); k chunks at kBase + half*8 and kBase + half*8 + 16
static __device__ __forceinline__ v16bf load_a_frag(const bf16_t* base, int row,
                                                    int kBase, int ld, int lane) {
  const bf16_t* p =
      base + (size_t)(row + (lane & 15)) * ld + kBase + ((lane >> 4) << 3);
  return combine8(*(const v8bf*)p, *(const v8bf*)(p + 16));
}

// B-fragment (32x16 bf16) from "column-major-by-k" matrix [col][ld]:
// lane: col = base + (l&15); k chunks at kBase + half*8 and kBase + half*8 + 16
static __device__ __forceinline__ v16bf load_b_frag(const bf16_t* base, int col,
                                                    int kBase, int ld, int lane) {
  const bf16_t* p =
      base + (size_t)(col + (lane & 15)) * ld + kBase + ((lane >> 4) << 3);
  return combine8(*(const v8bf*)p, *(const v8bf*)(p + 16));
}

static __device__ __forceinline__ v8f wmma_bf16(v16bf a, v16bf b, v8f c) {
  return __builtin_amdgcn_wmma_f32_16x16x32_bf16(false, a, false, b, (short)0, c,
                                                 false, false);
}

// max/sum across the 16 lanes sharing (lane>>4)
static __device__ __forceinline__ float grp16_max(float v) {
  v = fmaxf(v, __shfl_xor(v, 1, 32));
  v = fmaxf(v, __shfl_xor(v, 2, 32));
  v = fmaxf(v, __shfl_xor(v, 4, 32));
  v = fmaxf(v, __shfl_xor(v, 8, 32));
  return v;
}
static __device__ __forceinline__ float grp16_sum(float v) {
  v += __shfl_xor(v, 1, 32);
  v += __shfl_xor(v, 2, 32);
  v += __shfl_xor(v, 4, 32);
  v += __shfl_xor(v, 8, 32);
  return v;
}

// ---------------------------------------------------------------------------
// Elementwise conversions
// ---------------------------------------------------------------------------
__global__ void k_cvt_f32_bf16(const float* __restrict__ src,
                               bf16_t* __restrict__ dst, int n) {
  for (int i = blockIdx.x * blockDim.x + threadIdx.x; i < n;
       i += gridDim.x * blockDim.x)
    dst[i] = (bf16_t)src[i];
}

// src W[K][J] row-major f32 -> dst WT[J][K] bf16 (k contiguous per column)
__global__ void k_cvt_transpose(const float* __restrict__ W,
                                bf16_t* __restrict__ WT, int K, int J,
                                float scale) {
  int n = K * J;
  for (int i = blockIdx.x * blockDim.x + threadIdx.x; i < n;
       i += gridDim.x * blockDim.x) {
    int j = i / K, k = i - j * K;
    WT[i] = (bf16_t)(W[(size_t)k * J + j] * scale);
  }
}

// ---------------------------------------------------------------------------
// QKV projection: Y = X @ W. One wave -> 32x32 tile, K=1024 in steps of 32.
// mode (blockIdx.z): 0 = Q -> [b][h][n][64], 1 = K -> [b][h][m][64],
//                    2 = V -> transposed [b][h][d][m]
// ---------------------------------------------------------------------------
__global__ void k_proj_qkv(const bf16_t* __restrict__ xbf,
                           const bf16_t* __restrict__ ctxbf,
                           const bf16_t* __restrict__ WqT,
                           const bf16_t* __restrict__ WkT,
                           const bf16_t* __restrict__ WvT,
                           bf16_t* __restrict__ Qb, bf16_t* __restrict__ Kb,
                           bf16_t* __restrict__ VTb) {
  const int lane = threadIdx.x & 31;
  const int wave = (blockIdx.x * (blockDim.x >> 5)) + (threadIdx.x >> 5);
  const int mode = blockIdx.z;
  // 256 row tiles x 16 col tiles
  const int rt = wave >> 4;
  const int ct = wave & 15;
  const int rowBase = rt * 32;
  const int colBase = ct * 32;

  const bf16_t* X = (mode == 0) ? xbf : ctxbf;
  const bf16_t* WT = (mode == 0) ? WqT : (mode == 1) ? WkT : WvT;

  v8f c00 = {}, c01 = {}, c10 = {}, c11 = {};
#pragma unroll 4
  for (int kk = 0; kk < QD_; kk += 32) {
    v16bf a0 = load_a_frag(X, rowBase, kk, QD_, lane);
    v16bf a1 = load_a_frag(X, rowBase + 16, kk, QD_, lane);
    v16bf b0 = load_b_frag(WT, colBase, kk, QD_, lane);
    v16bf b1 = load_b_frag(WT, colBase + 16, kk, QD_, lane);
    c00 = wmma_bf16(a0, b0, c00);
    c01 = wmma_bf16(a0, b1, c01);
    c10 = wmma_bf16(a1, b0, c10);
    c11 = wmma_bf16(a1, b1, c11);
  }

  const int half = lane >> 4, lc = lane & 15;
  v8f cf[2][2] = {{c00, c01}, {c10, c11}};
#pragma unroll
  for (int i = 0; i < 2; ++i)
#pragma unroll
    for (int j = 0; j < 2; ++j)
#pragma unroll
      for (int r = 0; r < 8; ++r) {
        int row = rowBase + i * 16 + r + half * 8;  // flat b*2048 + n
        int col = colBase + j * 16 + lc;            // 0..511
        int bb = row >> 11, nn = row & (N_ - 1);
        int hh = col >> 6, dd = col & 63;
        bf16_t v = (bf16_t)cf[i][j][r];
        if (mode == 0)
          Qb[(((size_t)(bb * H_ + hh) * N_) + nn) * D_ + dd] = v;
        else if (mode == 1)
          Kb[(((size_t)(bb * H_ + hh) * M_) + nn) * D_ + dd] = v;
        else
          VTb[(((size_t)(bb * H_ + hh) * D_) + dd) * M_ + nn] = v;
      }
}

// ---------------------------------------------------------------------------
// Flash attention. One wave: 32 Q-rows of one (b,h); loop m in chunks of 32.
// Softmax scale pre-folded into Wq. Output bf16 [b][n][h*64+d].
// ---------------------------------------------------------------------------
__global__ void k_attention(const bf16_t* __restrict__ Qb,
                            const bf16_t* __restrict__ Kb,
                            const bf16_t* __restrict__ VTb,
                            bf16_t* __restrict__ AO) {
  __shared__ __align__(16) bf16_t plds[4][2][16 * 32];

  const int lane = threadIdx.x & 31;
  const int warp = threadIdx.x >> 5;
  const int wave = blockIdx.x * 4 + warp;
  const int bh = wave >> 6;            // 0..31
  const int nBase = (wave & 63) * 32;  // 0..2016
  const int b = bh >> 3, h = bh & 7;

  const bf16_t* qp = Qb + (size_t)bh * N_ * D_;
  const bf16_t* kp = Kb + (size_t)bh * M_ * D_;
  const bf16_t* vp = VTb + (size_t)bh * D_ * M_;

  v16bf q[2][2];
#pragma unroll
  for (int rt = 0; rt < 2; ++rt)
#pragma unroll
    for (int kc = 0; kc < 2; ++kc)
      q[rt][kc] = load_a_frag(qp, nBase + rt * 16, kc * 32, D_, lane);

  v8f o[2][4];
  float ms[2][8], ls[2][8];
#pragma unroll
  for (int rt = 0; rt < 2; ++rt) {
#pragma unroll
    for (int f = 0; f < 4; ++f) o[rt][f] = (v8f){};
#pragma unroll
    for (int r = 0; r < 8; ++r) { ms[rt][r] = -1e30f; ls[rt][r] = 0.f; }
  }

  const int half = lane >> 4, lc = lane & 15;

  for (int mb = 0; mb < M_; mb += 32) {
    v16bf kb[2][2];  // [m-col tile][d chunk]
#pragma unroll
    for (int mc = 0; mc < 2; ++mc)
#pragma unroll
      for (int dc = 0; dc < 2; ++dc)
        kb[mc][dc] = load_b_frag(kp, mb + mc * 16, dc * 32, D_, lane);
    v16bf vb[4];  // [d-col tile], K dim = m (32)
#pragma unroll
    for (int f = 0; f < 4; ++f)
      vb[f] = load_b_frag(vp, f * 16, mb, M_, lane);

#pragma unroll
    for (int rt = 0; rt < 2; ++rt) {
      v8f s0 = {}, s1 = {};
      s0 = wmma_bf16(q[rt][0], kb[0][0], s0);
      s0 = wmma_bf16(q[rt][1], kb[0][1], s0);
      s1 = wmma_bf16(q[rt][0], kb[1][0], s1);
      s1 = wmma_bf16(q[rt][1], kb[1][1], s1);

      float alpha[8];
#pragma unroll
      for (int r = 0; r < 8; ++r) {
        float rm = grp16_max(fmaxf(s0[r], s1[r]));
        float mnew = fmaxf(ms[rt][r], rm);
        alpha[r] = __expf(ms[rt][r] - mnew);
        ms[rt][r] = mnew;
      }
#pragma unroll
      for (int r = 0; r < 8; ++r) {
        s0[r] = __expf(s0[r] - ms[rt][r]);
        s1[r] = __expf(s1[r] - ms[rt][r]);
        float rs = grp16_sum(s0[r] + s1[r]);
        ls[rt][r] = ls[rt][r] * alpha[r] + rs;
      }
#pragma unroll
      for (int f = 0; f < 4; ++f)
#pragma unroll
        for (int r = 0; r < 8; ++r) o[rt][f][r] *= alpha[r];

      // C-layout -> A-layout via per-wave LDS bounce (in-order within wave)
      bf16_t* pl = plds[warp][rt];
#pragma unroll
      for (int r = 0; r < 8; ++r) {
        int row = r + half * 8;
        pl[row * 32 + lc] = (bf16_t)s0[r];
        pl[row * 32 + 16 + lc] = (bf16_t)s1[r];
      }
      const bf16_t* ap = pl + (lane & 15) * 32 + half * 8;
      v16bf pf = combine8(*(const v8bf*)ap, *(const v8bf*)(ap + 16));

#pragma unroll
      for (int f = 0; f < 4; ++f) o[rt][f] = wmma_bf16(pf, vb[f], o[rt][f]);
    }
  }

#pragma unroll
  for (int rt = 0; rt < 2; ++rt)
#pragma unroll
    for (int f = 0; f < 4; ++f)
#pragma unroll
      for (int r = 0; r < 8; ++r) {
        int n = nBase + rt * 16 + r + half * 8;
        int col = h * 64 + f * 16 + lc;
        AO[((size_t)(b * N_ + n)) * INNER_ + col] =
            (bf16_t)(o[rt][f][r] / ls[rt][r]);
      }
}

// ---------------------------------------------------------------------------
// Out projection: out = AO @ Wo + bo  (f32 output). 32x32 tile per wave.
// ---------------------------------------------------------------------------
__global__ void k_out_proj(const bf16_t* __restrict__ AO,
                           const bf16_t* __restrict__ WoT,
                           const float* __restrict__ bo,
                           float* __restrict__ out) {
  const int lane = threadIdx.x & 31;
  const int wave = blockIdx.x * (blockDim.x >> 5) + (threadIdx.x >> 5);
  const int rt = wave >> 5;        // 256 row tiles
  const int ct = wave & 31;        // 32 col tiles
  const int rowBase = rt * 32, colBase = ct * 32;

  v8f c00 = {}, c01 = {}, c10 = {}, c11 = {};
#pragma unroll 4
  for (int kk = 0; kk < INNER_; kk += 32) {
    v16bf a0 = load_a_frag(AO, rowBase, kk, INNER_, lane);
    v16bf a1 = load_a_frag(AO, rowBase + 16, kk, INNER_, lane);
    v16bf b0 = load_b_frag(WoT, colBase, kk, INNER_, lane);
    v16bf b1 = load_b_frag(WoT, colBase + 16, kk, INNER_, lane);
    c00 = wmma_bf16(a0, b0, c00);
    c01 = wmma_bf16(a0, b1, c01);
    c10 = wmma_bf16(a1, b0, c10);
    c11 = wmma_bf16(a1, b1, c11);
  }
  const int half = lane >> 4, lc = lane & 15;
  v8f cf[2][2] = {{c00, c01}, {c10, c11}};
#pragma unroll
  for (int j = 0; j < 2; ++j) {
    int col = colBase + j * 16 + lc;
    float bias = bo[col];
#pragma unroll
    for (int i = 0; i < 2; ++i)
#pragma unroll
      for (int r = 0; r < 8; ++r) {
        int row = rowBase + i * 16 + r + half * 8;
        out[(size_t)row * QD_ + col] = cf[i][j][r] + bias;
      }
  }
}

// ---------------------------------------------------------------------------
// Host launch
// ---------------------------------------------------------------------------
extern "C" void kernel_launch(void* const* d_in, const int* in_sizes, int n_in,
                              void* d_out, int out_size, void* d_ws,
                              size_t ws_size, hipStream_t stream) {
  const float* x   = (const float*)d_in[0];
  const float* ctx = (const float*)d_in[1];
  const float* Wq  = (const float*)d_in[2];
  const float* Wk  = (const float*)d_in[3];
  const float* Wv  = (const float*)d_in[4];
  const float* Wo  = (const float*)d_in[5];
  const float* bo  = (const float*)d_in[6];
  float* out = (float*)d_out;

  size_t off = 0;
  auto alloc = [&](size_t bytes) {
    void* p = (char*)d_ws + off;
    off += (bytes + 255) & ~(size_t)255;
    return p;
  };
  bf16_t* xbf   = (bf16_t*)alloc((size_t)ROWS_ * QD_ * 2);
  bf16_t* ctxbf = (bf16_t*)alloc((size_t)ROWS_ * QD_ * 2);
  bf16_t* WqT   = (bf16_t*)alloc((size_t)INNER_ * QD_ * 2);
  bf16_t* WkT   = (bf16_t*)alloc((size_t)INNER_ * QD_ * 2);
  bf16_t* WvT   = (bf16_t*)alloc((size_t)INNER_ * QD_ * 2);
  bf16_t* WoT   = (bf16_t*)alloc((size_t)QD_ * INNER_ * 2);
  bf16_t* Qb    = (bf16_t*)alloc((size_t)B_ * H_ * N_ * D_ * 2);
  bf16_t* Kb    = (bf16_t*)alloc((size_t)B_ * H_ * M_ * D_ * 2);
  bf16_t* VTb   = (bf16_t*)alloc((size_t)B_ * H_ * D_ * M_ * 2);
  bf16_t* AO    = (bf16_t*)alloc((size_t)ROWS_ * INNER_ * 2);

  const float scale = 0.125f;  // 64^-0.5, folded into Wq

  k_cvt_f32_bf16<<<4096, 256, 0, stream>>>(x, xbf, ROWS_ * QD_);
  k_cvt_f32_bf16<<<4096, 256, 0, stream>>>(ctx, ctxbf, ROWS_ * QD_);
  k_cvt_transpose<<<1024, 256, 0, stream>>>(Wq, WqT, QD_, INNER_, scale);
  k_cvt_transpose<<<1024, 256, 0, stream>>>(Wk, WkT, QD_, INNER_, 1.0f);
  k_cvt_transpose<<<1024, 256, 0, stream>>>(Wv, WvT, QD_, INNER_, 1.0f);
  k_cvt_transpose<<<1024, 256, 0, stream>>>(Wo, WoT, INNER_, QD_, 1.0f);

  // 4096 waves per mode, 8 waves/block -> 512 blocks, z = {Q,K,V}
  k_proj_qkv<<<dim3(512, 1, 3), 256, 0, stream>>>(xbf, ctxbf, WqT, WkT, WvT,
                                                  Qb, Kb, VTb);
  // 2048 waves, 4 waves/block -> 512 blocks
  k_attention<<<512, 128, 0, stream>>>(Qb, Kb, VTb, AO);
  // 8192 waves, 8 waves/block -> 1024 blocks
  k_out_proj<<<1024, 256, 0, stream>>>(AO, WoT, bo, out);
}
